// RMTMemory_57724360458758
// MI455X (gfx1250) — compile-verified, matrix operand order checked by reading, not run on hardware
//
#include <hip/hip_runtime.h>
#include <hip/hip_bf16.h>

// ---------------------------------------------------------------------------
// RMT memory-transformer forward for MI455X (gfx1250), bf16 WMMA everywhere.
// ---------------------------------------------------------------------------

constexpr int KD   = 2560;   // model dim
constexpr int KM   = 128;    // memory slots
constexpr int KSEG = 1024;   // segment length
constexpr int KS   = 4096;   // full sequence
constexpr int KB   = 2;      // batch
constexpr int KH   = 8;      // heads
constexpr int KHD  = 320;    // head dim

typedef __attribute__((ext_vector_type(16))) __bf16 v16bf;
typedef __attribute__((ext_vector_type(8)))  float  v8f;

union FragB {
    v16bf v;
    uint4 q[2];
};

__device__ __forceinline__ unsigned short f2bf(float f) {
    unsigned u = __builtin_bit_cast(unsigned, f);
    unsigned r = u + 0x7FFFu + ((u >> 16) & 1u);   // round-to-nearest-even
    return (unsigned short)(r >> 16);
}

__device__ __forceinline__ FragB load_frag(const unsigned short* p) {
    FragB f;
    f.q[0] = *(const uint4*)p;          // K = kb .. kb+7
    f.q[1] = *(const uint4*)(p + 16);   // K = kb+16 .. kb+23
    return f;
}

__device__ __forceinline__ v8f wmma_bf16(const FragB& a, const FragB& b, v8f c) {
    return __builtin_amdgcn_wmma_f32_16x16x32_bf16(
        false, a.v, false, b.v, (short)0, c, false, false);
}

__device__ __forceinline__ float sigmoidf_(float x) {
    return 1.0f / (1.0f + __expf(-x));
}

// --- CDNA5 async global->LDS copy (bypasses VGPRs, tracked by ASYNCcnt) ----
// Generic (flat) pointers into LDS have the LDS byte offset in addr[31:0]
// (ISA 10.2 aperture mapping), so truncation yields the DS address.
__device__ __forceinline__ void async_g2l_b128(void* lds, const void* g) {
    unsigned l = (unsigned)(size_t)lds;
    asm volatile("global_load_async_to_lds_b128 %0, %1, off"
                 :: "v"(l), "v"(g) : "memory");
}
__device__ __forceinline__ void wait_asynccnt0() {
    asm volatile("s_wait_asynccnt 0x0" ::: "memory");
}

// ---------------------------------------------------------------------------
// Tiled GEMM:  C[M,N] = A[M,K] * Bw[K,N] (+bias) (+C)
//   BM=BN=128, BK=32, 256 threads (8 waves), wave = 2x4 grid of 16x16 tiles.
//   A fp32: converted to bf16 in VGPRs while staging to LDS.
//   A bf16: staged with global_load_async_to_lds_b128 (no VGPR round trip).
//   Bw is fp32 weights (row-major KxN), staged transposed (col-major) in LDS.
// ---------------------------------------------------------------------------
template<bool A_F32, bool OUT_F32, bool OUT_BF, bool HAS_BIAS, bool ACC>
__global__ __launch_bounds__(256, 2)
void gemm_wmma_bf16(const void* __restrict__ Ap, long long strideA, int lda,
                    const float* __restrict__ Bw, int ldb,
                    float* __restrict__ Cf, unsigned short* __restrict__ Cb,
                    long long strideC, int ldc,
                    const float* __restrict__ bias, int Kdim)
{
    __shared__ __align__(16) unsigned short As[128 * 40];  // row-major, stride 40
    __shared__ __align__(16) unsigned short Bs[128 * 40];  // col-major, stride 40

    const int tid  = threadIdx.x;
    const int wave = tid >> 5, lane = tid & 31;
    const int li = lane & 15, lh = lane >> 4, kb = lh * 8;
    const int wr = wave >> 1, wc = wave & 1;
    const int bz = blockIdx.z;
    const int rowBlock = blockIdx.y * 128;
    const int colBlock = blockIdx.x * 128;
    const int am = tid & 127;           // coop row (A) / col (B)
    const int ak = (tid >> 7) * 16;     // coop K offset (0 or 16)

    v8f acc[2][4] = {};

    const float*          Af = (const float*)Ap + (size_t)bz * strideA;
    const unsigned short* Ah = (const unsigned short*)Ap + (size_t)bz * strideA;

#pragma unroll 1
    for (int k0 = 0; k0 < Kdim; k0 += 32) {
        // ---- stage A tile ----
        if constexpr (A_F32) {
            const float4* ap = (const float4*)(Af + (size_t)(rowBlock + am) * lda + k0 + ak);
            float4 f0 = ap[0], f1 = ap[1], f2 = ap[2], f3 = ap[3];
            uint4 w0, w1;
            w0.x = (unsigned)f2bf(f0.x) | ((unsigned)f2bf(f0.y) << 16);
            w0.y = (unsigned)f2bf(f0.z) | ((unsigned)f2bf(f0.w) << 16);
            w0.z = (unsigned)f2bf(f1.x) | ((unsigned)f2bf(f1.y) << 16);
            w0.w = (unsigned)f2bf(f1.z) | ((unsigned)f2bf(f1.w) << 16);
            w1.x = (unsigned)f2bf(f2.x) | ((unsigned)f2bf(f2.y) << 16);
            w1.y = (unsigned)f2bf(f2.z) | ((unsigned)f2bf(f2.w) << 16);
            w1.z = (unsigned)f2bf(f3.x) | ((unsigned)f2bf(f3.y) << 16);
            w1.w = (unsigned)f2bf(f3.z) | ((unsigned)f2bf(f3.w) << 16);
            *(uint4*)&As[am * 40 + ak]     = w0;
            *(uint4*)&As[am * 40 + ak + 8] = w1;
        } else {
            // bf16 activations: DMA straight into LDS, no VGPR staging.
            const unsigned short* ap = Ah + (size_t)(rowBlock + am) * lda + k0 + ak;
            async_g2l_b128(&As[am * 40 + ak],     ap);
            async_g2l_b128(&As[am * 40 + ak + 8], ap + 8);
        }
        // ---- stage B tile transposed (weights fp32 -> bf16) ----
        {
            const float* bp = Bw + (size_t)(k0 + ak) * ldb + colBlock + am;
#pragma unroll
            for (int jj = 0; jj < 16; jj += 2) {
                unsigned lo = f2bf(bp[(size_t)jj * ldb]);
                unsigned hi = f2bf(bp[(size_t)(jj + 1) * ldb]);
                *(unsigned*)&Bs[am * 40 + ak + jj] = lo | (hi << 16);
            }
        }
        if constexpr (!A_F32) wait_asynccnt0();   // A tile landed in LDS
        __syncthreads();

        if (k0 + 32 < Kdim) {   // prefetch next tiles (global_prefetch_b8)
            if constexpr (A_F32)
                __builtin_prefetch(Af + (size_t)(rowBlock + am) * lda + k0 + 32 + ak, 0, 1);
            else
                __builtin_prefetch(Ah + (size_t)(rowBlock + am) * lda + k0 + 32 + ak, 0, 1);
            __builtin_prefetch(Bw + (size_t)(k0 + 32 + ak) * ldb + colBlock + am, 0, 1);
        }

        FragB fa[2], fb[4];
#pragma unroll
        for (int rt = 0; rt < 2; rt++)
            fa[rt] = load_frag(&As[(wr * 32 + rt * 16 + li) * 40 + kb]);
#pragma unroll
        for (int ct = 0; ct < 4; ct++)
            fb[ct] = load_frag(&Bs[(wc * 64 + ct * 16 + li) * 40 + kb]);
#pragma unroll
        for (int rt = 0; rt < 2; rt++)
#pragma unroll
            for (int ct = 0; ct < 4; ct++)
                acc[rt][ct] = wmma_bf16(fa[rt], fb[ct], acc[rt][ct]);
        __syncthreads();
    }

    float*          Cfb = Cf + (size_t)bz * strideC;
    unsigned short* Cbb = Cb + (size_t)bz * strideC;
#pragma unroll
    for (int rt = 0; rt < 2; rt++) {
#pragma unroll
        for (int ct = 0; ct < 4; ct++) {
            const int col  = colBlock + wc * 64 + ct * 16 + li;
            const int row0 = rowBlock + wr * 32 + rt * 16 + lh * 8;
#pragma unroll
            for (int r = 0; r < 8; r++) {
                const size_t idx = (size_t)(row0 + r) * ldc + col;
                float v = acc[rt][ct][r];
                if constexpr (ACC)      v += Cfb[idx];
                if constexpr (HAS_BIAS) v += bias[col];
                if constexpr (OUT_F32)  Cfb[idx] = v;
                if constexpr (OUT_BF)   Cbb[idx] = f2bf(v);
            }
        }
    }
}

// ---------------------------------------------------------------------------
// Read attention: 1024 queries over 128 memory slots, per (b,h) block.
//   q,k,v,ctx are bf16 (B,*,D) with per-head slice of KHD=320 dims.
//   LDS: v^T (320x128) staged once; per-wave score/P buffers.
//   ks / j loops kept rolled (#pragma unroll 1) to bound VGPR pressure.
// ---------------------------------------------------------------------------
__global__ __launch_bounds__(256, 1)
void attn_read_kernel(const unsigned short* __restrict__ Q,
                      const unsigned short* __restrict__ Kk,
                      const unsigned short* __restrict__ V,
                      unsigned short* __restrict__ Ctx)
{
    extern __shared__ char smem_raw[];
    unsigned short* vt = (unsigned short*)smem_raw;                         // 320 x 136
    float*          sc = (float*)(smem_raw + 320 * 136 * 2);                // 8 x 16 x 128
    unsigned short* pb = (unsigned short*)((char*)sc + 8 * 16 * 128 * 4);   // 8 x 16 x 128

    const int h = blockIdx.x, b = blockIdx.y;
    const int tid = threadIdx.x, wave = tid >> 5, lane = tid & 31;
    const int li = lane & 15, lh = lane >> 4, kb = lh * 8;
    const float scale = 0.05590169943749474f;   // 1/sqrt(320)

    // stage v^T into LDS (dim-major, slot-contiguous)
    for (int idx = tid; idx < KM * KHD; idx += 256) {
        int slot = idx / KHD, dim = idx - slot * KHD;
        vt[dim * 136 + slot] = V[(size_t)(b * KM + slot) * KD + h * KHD + dim];
    }
    __syncthreads();

    float*          myS = sc + wave * 16 * 128;
    unsigned short* myP = pb + wave * 16 * 128;

#pragma unroll 1
    for (int j = 0; j < 8; j++) {               // 8 q-tiles (16 rows) per wave
        const int qrow0 = (wave + j * 8) * 16;

        // ---- scores = q x k^T ----
        v8f s[8] = {};
#pragma unroll 1
        for (int ks = 0; ks < 10; ks++) {
            FragB fa = load_frag(Q + (size_t)(b * KSEG + qrow0 + li) * KD + h * KHD + ks * 32 + kb);
#pragma unroll
            for (int nt = 0; nt < 8; nt++) {
                FragB fk = load_frag(Kk + (size_t)(b * KM + nt * 16 + li) * KD + h * KHD + ks * 32 + kb);
                s[nt] = wmma_bf16(fa, fk, s[nt]);
            }
        }
#pragma unroll
        for (int nt = 0; nt < 8; nt++)
#pragma unroll
            for (int r = 0; r < 8; r++)
                myS[(lh * 8 + r) * 128 + nt * 16 + li] = s[nt][r] * scale;
        __syncthreads();

        // ---- softmax per row (lanes 0..15 each own one row) ----
        if (lane < 16) {
            const int row = lane;
            float m = -1e30f;
            for (int c = 0; c < 128; c++) m = fmaxf(m, myS[row * 128 + c]);
            float sum = 0.0f;
            for (int c = 0; c < 128; c++) {
                float e = __expf(myS[row * 128 + c] - m);
                sum += e;
                myS[row * 128 + c] = e;
            }
            float inv = 1.0f / sum;
            for (int c = 0; c < 128; c++)
                myP[row * 128 + c] = f2bf(myS[row * 128 + c] * inv);
        }
        __syncthreads();

        // ---- out = P x V ----
        FragB pf[4];
#pragma unroll
        for (int kc = 0; kc < 4; kc++)
            pf[kc] = load_frag(myP + li * 128 + kc * 32 + kb);
#pragma unroll 1
        for (int nt = 0; nt < 20; nt++) {
            v8f o = {};
#pragma unroll
            for (int kc = 0; kc < 4; kc++) {
                FragB fv = load_frag(vt + (nt * 16 + li) * 136 + kc * 32 + kb);
                o = wmma_bf16(pf[kc], fv, o);
            }
#pragma unroll
            for (int r = 0; r < 8; r++)
                Ctx[(size_t)(b * KSEG + qrow0 + lh * 8 + r) * KD + h * KHD + nt * 16 + li] =
                    f2bf(o[r]);
        }
        __syncthreads();
    }
}

// ---------------------------------------------------------------------------
// Write attention (flash-style): 128 queries over 1024 keys, per (b,h) block.
//   512 threads / 16 waves: wave pair p owns q-tile p (16 rows); the even
//   wave computes the 16x32 score tile + online softmax; both waves of the
//   pair do P x V for their own half (160 dims -> O[10] = 80 VGPRs, no spill).
// ---------------------------------------------------------------------------
__global__ __launch_bounds__(512, 1)
void attn_write_kernel(const unsigned short* __restrict__ Qw,   // (KM, D), batch-shared
                       const unsigned short* __restrict__ Kw,   // (B, SEG, D)
                       const unsigned short* __restrict__ Vw,   // (B, SEG, D)
                       unsigned short* __restrict__ Out)        // (B, KM, D)
{
    extern __shared__ char smem_raw[];
    unsigned short* vtc   = (unsigned short*)smem_raw;                        // 320 x 40
    float*          sc    = (float*)(smem_raw + 320 * 40 * 2);                // 8 x 16 x 32
    unsigned short* pbuf  = (unsigned short*)((char*)sc + 8 * 16 * 32 * 4);   // 8 x 16 x 32
    float*          stats = (float*)((char*)pbuf + 8 * 16 * 32 * 2);          // 8 x 48 (m,l,alpha)

    const int h = blockIdx.x, b = blockIdx.y;
    const int tid = threadIdx.x, wave = tid >> 5, lane = tid & 31;
    const int li = lane & 15, lh = lane >> 4, kb = lh * 8;
    const int qt = wave >> 1;          // q-tile (0..7)
    const int half = wave & 1;         // output-dim half (0..1)
    const float scale = 0.05590169943749474f;

    float*          myS = sc + qt * 16 * 32;
    unsigned short* myP = pbuf + qt * 16 * 32;
    float*          myM = stats + qt * 48;

    if (half == 0 && lane < 16) { myM[lane] = -1e30f; myM[16 + lane] = 0.0f; }
    v8f O[10] = {};

#pragma unroll 1
    for (int kc = 0; kc < KSEG / 32; kc++) {
        __syncthreads();   // protect vtc/P from previous iteration's readers
        for (int idx = tid; idx < KHD * 32; idx += 512) {
            int slot = idx / KHD, dim = idx - slot * KHD;
            vtc[dim * 40 + slot] =
                Vw[(size_t)(b * KSEG + kc * 32 + slot) * KD + h * KHD + dim];
        }
        __syncthreads();

        // ---- scores for this key chunk (even wave of each pair only) ----
        if (half == 0) {
            v8f s[2] = {};
#pragma unroll 1
            for (int ks = 0; ks < 10; ks++) {
                FragB fa = load_frag(Qw + (size_t)(qt * 16 + li) * KD + h * KHD + ks * 32 + kb);
#pragma unroll
                for (int nt = 0; nt < 2; nt++) {
                    FragB fk = load_frag(
                        Kw + (size_t)(b * KSEG + kc * 32 + nt * 16 + li) * KD + h * KHD + ks * 32 + kb);
                    s[nt] = wmma_bf16(fa, fk, s[nt]);
                }
            }
#pragma unroll
            for (int nt = 0; nt < 2; nt++)
#pragma unroll
                for (int r = 0; r < 8; r++)
                    myS[(lh * 8 + r) * 32 + nt * 16 + li] = s[nt][r] * scale;
        }
        __syncthreads();

        // ---- online softmax update (even wave, lanes 0..15, one row each) ----
        if (half == 0 && lane < 16) {
            const int row = lane;
            float mo = myM[row], lo = myM[16 + row];
            float mx = mo;
            for (int c = 0; c < 32; c++) mx = fmaxf(mx, myS[row * 32 + c]);
            float alpha = __expf(mo - mx);
            float sum = 0.0f;
            for (int c = 0; c < 32; c++) {
                float e = __expf(myS[row * 32 + c] - mx);
                sum += e;
                myP[row * 32 + c] = f2bf(e);
            }
            myM[row] = mx;
            myM[16 + row] = lo * alpha + sum;
            myM[32 + row] = alpha;
        }
        __syncthreads();

        // ---- rescale accumulators, then O += P x V_chunk (both waves) ----
        float al[8];
#pragma unroll
        for (int r = 0; r < 8; r++) al[r] = myM[32 + lh * 8 + r];
#pragma unroll
        for (int nt = 0; nt < 10; nt++)
#pragma unroll
            for (int r = 0; r < 8; r++)
                O[nt][r] *= al[r];

        FragB pf = load_frag(myP + li * 32 + kb);
#pragma unroll
        for (int nt = 0; nt < 10; nt++) {
            FragB fv = load_frag(vtc + ((half * 10 + nt) * 16 + li) * 40 + kb);
            O[nt] = wmma_bf16(pf, fv, O[nt]);
        }
    }
    __syncthreads();

    float linv[8];
#pragma unroll
    for (int r = 0; r < 8; r++) linv[r] = 1.0f / myM[16 + lh * 8 + r];
#pragma unroll
    for (int nt = 0; nt < 10; nt++)
#pragma unroll
        for (int r = 0; r < 8; r++)
            Out[(size_t)(b * KM + qt * 16 + lh * 8 + r) * KD + h * KHD +
                (half * 10 + nt) * 16 + li] = f2bf(O[nt][r] * linv[r]);
}

// ---------------------------------------------------------------------------
// Fused epilogues
// ---------------------------------------------------------------------------
__global__ void h_epilogue_kernel(const float* __restrict__ hid,
                                  const float* __restrict__ glin,
                                  const float* __restrict__ ctxo,
                                  float* __restrict__ out_h,
                                  unsigned short* __restrict__ h_bf,
                                  int segOff, int n)
{
    int j = blockIdx.x * blockDim.x + threadIdx.x;
    if (j >= n) return;
    const int bsz = KSEG * KD;
    int b = j / bsz, r = j - b * bsz;
    size_t gidx = (size_t)b * KS * KD + segOff + r;
    float hv = hid[gidx] + sigmoidf_(glin[j]) * ctxo[j];
    out_h[gidx] = hv;
    h_bf[j] = f2bf(hv);
}

__global__ void mem_update_kernel(const float* __restrict__ glin,
                                  const float* __restrict__ newm,
                                  float* __restrict__ memf,
                                  unsigned short* __restrict__ membf,
                                  int n)
{
    int j = blockIdx.x * blockDim.x + threadIdx.x;
    if (j >= n) return;
    float g = sigmoidf_(glin[j]);
    float m = g * newm[j] + (1.0f - g) * memf[j];
    memf[j] = m;
    membf[j] = f2bf(m);
}

// ---------------------------------------------------------------------------
// Host orchestration
// ---------------------------------------------------------------------------
extern "C" void kernel_launch(void* const* d_in, const int* in_sizes, int n_in,
                              void* d_out, int out_size, void* d_ws, size_t ws_size,
                              hipStream_t stream)
{
    const float* hidden   = (const float*)d_in[0];
    const float* init_mem = (const float*)d_in[1];
    const float* wq_r     = (const float*)d_in[2];
    const float* wk_r     = (const float*)d_in[3];
    const float* wv_r     = (const float*)d_in[4];
    const float* wo_r     = (const float*)d_in[5];
    const float* wg_r     = (const float*)d_in[6];
    const float* bg_r     = (const float*)d_in[7];
    const float* wrt_q    = (const float*)d_in[8];
    const float* wq_w     = (const float*)d_in[9];
    const float* wk_w     = (const float*)d_in[10];
    const float* wv_w     = (const float*)d_in[11];
    const float* wo_w     = (const float*)d_in[12];
    const float* wg_w     = (const float*)d_in[13];
    const float* bg_w     = (const float*)d_in[14];
    float* out = (float*)d_out;

    const size_t EL_ACT = (size_t)KB * KSEG * KD;   // 5,242,880
    const size_t EL_MEM = (size_t)KB * KM * KD;     //   655,360

    char* wp = (char*)d_ws;
    auto carve = [&](size_t bytes) {
        char* p = wp;
        wp += (bytes + 255) & ~(size_t)255;
        return p;
    };
    unsigned short* q_bf   = (unsigned short*)carve(EL_ACT * 2);
    unsigned short* kr_bf  = (unsigned short*)carve(EL_MEM * 2);
    unsigned short* vr_bf  = (unsigned short*)carve(EL_MEM * 2);
    unsigned short* ctx_bf = (unsigned short*)carve(EL_ACT * 2);
    float*          glin_r = (float*)carve(EL_ACT * 4);
    float*          ctxo   = (float*)carve(EL_ACT * 4);
    unsigned short* h_bf   = (unsigned short*)carve(EL_ACT * 2);
    unsigned short* kw_bf  = (unsigned short*)carve(EL_ACT * 2);
    unsigned short* vw_bf  = (unsigned short*)carve(EL_ACT * 2);
    unsigned short* mqw_bf = (unsigned short*)carve((size_t)KM * KD * 2);
    unsigned short* aw_bf  = (unsigned short*)carve(EL_MEM * 2);
    float*          memf   = (float*)carve(EL_MEM * 4);
    unsigned short* membf  = (unsigned short*)carve(EL_MEM * 2);
    float*          newmf  = (float*)carve(EL_MEM * 4);
    float*          glin_w = (float*)carve(EL_MEM * 4);

    const dim3 blk(256);
    const dim3 blkAW(512);
    const dim3 gAct(KD / 128, KSEG / 128, KB);   // (20, 8, 2)
    const dim3 gMem(KD / 128, KM / 128, KB);     // (20, 1, 2)
    const dim3 gMq (KD / 128, KM / 128, 1);
    const dim3 gAttn(KH, KB);

    const int SM_AR = 320 * 136 * 2 + 8 * 16 * 128 * 4 + 8 * 16 * 128 * 2;  // 185,344 B
    const int SM_AW = 320 * 40 * 2 + 8 * 16 * 32 * 4 + 8 * 16 * 32 * 2 + 8 * 48 * 4;

    const long long sAct = (long long)KSEG * KD;
    const long long sMem = (long long)KM * KD;
    const long long sHid = (long long)KS * KD;

    // mq = write_queries @ wq_w (batch-shared)
    gemm_wmma_bf16<true, false, true, false, false><<<gMq, blk, 0, stream>>>(
        wrt_q, 0, KD, wq_w, KD, nullptr, mqw_bf, 0, KD, nullptr, KD);

    for (int i = 0; i < 4; i++) {
        const float* segA = hidden + (size_t)i * KSEG * KD;

        // q = seg @ wq_r
        gemm_wmma_bf16<true, false, true, false, false><<<gAct, blk, 0, stream>>>(
            segA, sHid, KD, wq_r, KD, nullptr, q_bf, sAct, KD, nullptr, KD);
        // k/v = mem @ wk_r / wv_r  (segment 0: broadcast initial_memory)
        const void* memA = (i == 0) ? (const void*)init_mem : (const void*)memf;
        const long long memS = (i == 0) ? 0 : sMem;
        gemm_wmma_bf16<true, false, true, false, false><<<gMem, blk, 0, stream>>>(
            memA, memS, KD, wk_r, KD, nullptr, kr_bf, sMem, KD, nullptr, KD);
        gemm_wmma_bf16<true, false, true, false, false><<<gMem, blk, 0, stream>>>(
            memA, memS, KD, wv_r, KD, nullptr, vr_bf, sMem, KD, nullptr, KD);

        attn_read_kernel<<<gAttn, blk, SM_AR, stream>>>(q_bf, kr_bf, vr_bf, ctx_bf);

        // gate_lin = seg @ wg_r + bg_r ;  ctxo = ctx @ wo_r
        gemm_wmma_bf16<true, true, false, true, false><<<gAct, blk, 0, stream>>>(
            segA, sHid, KD, wg_r, KD, glin_r, nullptr, sAct, KD, bg_r, KD);
        gemm_wmma_bf16<false, true, false, false, false><<<gAct, blk, 0, stream>>>(
            ctx_bf, sAct, KD, wo_r, KD, ctxo, nullptr, sAct, KD, nullptr, KD);

        const int nAct = (int)EL_ACT;
        h_epilogue_kernel<<<(nAct + 255) / 256, blk, 0, stream>>>(
            hidden, glin_r, ctxo, out, h_bf, i * KSEG * KD, nAct);

        // k/v for write attention: h @ wk_w / wv_w
        gemm_wmma_bf16<false, false, true, false, false><<<gAct, blk, 0, stream>>>(
            h_bf, sAct, KD, wk_w, KD, nullptr, kw_bf, sAct, KD, nullptr, KD);
        gemm_wmma_bf16<false, false, true, false, false><<<gAct, blk, 0, stream>>>(
            h_bf, sAct, KD, wv_w, KD, nullptr, vw_bf, sAct, KD, nullptr, KD);

        attn_write_kernel<<<gAttn, blkAW, SM_AW, stream>>>(mqw_bf, kw_bf, vw_bf, aw_bf);

        if (i == 0) {
            // memory = attn_out @ wo_w
            gemm_wmma_bf16<false, true, true, false, false><<<gMem, blk, 0, stream>>>(
                aw_bf, sMem, KD, wo_w, KD, memf, membf, sMem, KD, nullptr, KD);
        } else {
            gemm_wmma_bf16<false, true, false, false, false><<<gMem, blk, 0, stream>>>(
                aw_bf, sMem, KD, wo_w, KD, newmf, nullptr, sMem, KD, nullptr, KD);
            // g_lin = mem @ wg_w[:D] + new @ wg_w[D:] + bg_w
            gemm_wmma_bf16<true, true, false, true, false><<<gMem, blk, 0, stream>>>(
                memf, sMem, KD, wg_w, KD, glin_w, nullptr, sMem, KD, bg_w, KD);
            gemm_wmma_bf16<true, true, false, false, true><<<gMem, blk, 0, stream>>>(
                newmf, sMem, KD, wg_w + (size_t)KD * KD, KD, glin_w, nullptr, sMem, KD,
                nullptr, KD);
            const int nMem = (int)EL_MEM;
            mem_update_kernel<<<(nMem + 255) / 256, blk, 0, stream>>>(
                glin_w, newmf, memf, membf, nMem);
        }
    }
}
